// VAE_47536698032623
// MI455X (gfx1250) — matrix operand order repaired and checked by reading, not compiled, not run on hardware
//
#include <hip/hip_runtime.h>
#include <hip/hip_bf16.h>

typedef __bf16 bf16_t;
typedef __attribute__((ext_vector_type(16))) __bf16 v16bf;
typedef __attribute__((ext_vector_type(8)))  float  v8f;

__device__ __forceinline__ float sigf(float x) { return 1.0f / (1.0f + __expf(-x)); }

// B: 32x16 bf16 fragment, B = W^T tile, W row-major (N,K). One 32B contiguous load.
// lane n (<16): K = k..k+15 of column n; lane n+16: K = k+16..k+31
__device__ __forceinline__ v16bf load_b(const bf16_t* __restrict__ W, int ldw,
                                        int nBase, int k, int lane) {
  int col = lane & 15;
  int kb = (lane >> 4) * 16;
  return *(const v16bf*)(W + (size_t)(nBase + col) * ldw + k + kb);
}

// A fragment from a bf16 row pointer (LDS): elements {0..15}.
__device__ __forceinline__ v16bf load_a_row(const __bf16* __restrict__ p) {
  v16bf a;
#pragma unroll
  for (int j = 0; j < 8; ++j) a[j] = p[j];
#pragma unroll
  for (int j = 0; j < 8; ++j) a[8 + j] = p[8 + j];
  return a;
}

// K-chain for one weight source, unrolled x4: issue 4 B global loads + 8 A ds
// loads up front, then 4 chained WMMAs -> overlaps L2 latency with matrix work.
// K must be a multiple of 128 (all call sites: 128/512/1024).
__device__ __forceinline__ v8f k_chain(v8f acc, const __bf16* __restrict__ aS, int KT,
                                       int kOff, int K, const bf16_t* __restrict__ W,
                                       int ldw, int colBase, int lane) {
  const bf16_t* wRow = W + (size_t)(colBase + (lane & 15)) * ldw + (lane >> 4) * 16;
  const __bf16* aRow = aS + (size_t)(lane & 15) * KT + kOff + (lane >> 4) * 16;
  for (int k = 0; k < K; k += 128) {
    v16bf b0 = *(const v16bf*)(wRow + k);
    v16bf b1 = *(const v16bf*)(wRow + k + 32);
    v16bf b2 = *(const v16bf*)(wRow + k + 64);
    v16bf b3 = *(const v16bf*)(wRow + k + 96);
    v16bf a0 = load_a_row(aRow + k);
    v16bf a1 = load_a_row(aRow + k + 32);
    v16bf a2 = load_a_row(aRow + k + 64);
    v16bf a3 = load_a_row(aRow + k + 96);
    acc = __builtin_amdgcn_wmma_f32_16x16x32_bf16(false, a0, false, b0, (short)0, acc, false, false);
    acc = __builtin_amdgcn_wmma_f32_16x16x32_bf16(false, a1, false, b1, (short)0, acc, false, false);
    acc = __builtin_amdgcn_wmma_f32_16x16x32_bf16(false, a2, false, b2, (short)0, acc, false, false);
    acc = __builtin_amdgcn_wmma_f32_16x16x32_bf16(false, a3, false, b3, (short)0, acc, false, false);
  }
  return acc;
}

// ---- Generic WMMA GEMM: Y = act( A1@W1^T [+ A2@W2^T] + bias [+ Yadd] ) -----
// One block = one 16-row stripe x 8 column tiles (128 cols). The f32 activation
// stripe (A1|A2 concatenated along K) is converted to bf16 ONCE into LDS by all
// 256 threads, then each of the 8 waves runs the K-chains of WMMAs reading A
// from LDS (ds_load_b128) and B from L2-resident bf16 weights (global 32B/lane).
// grid.x = (M/16) * (N/128); dynamic LDS = 16*(K1+K2)*2 bytes.
__global__ void k_gemm(float* __restrict__ Y, int ldy,
                       float* __restrict__ Y2, long long ldy2,
                       const float* __restrict__ Yadd, int ldadd,
                       const float* __restrict__ A1, int lda1, int K1,
                       const bf16_t* __restrict__ W1, int ldw1,
                       const float* __restrict__ A2, int lda2, int K2,
                       const bf16_t* __restrict__ W2, int ldw2,
                       const float* __restrict__ bias,
                       int N, int act, int a1_swz) {
  extern __shared__ char smem[];
  __bf16* aS = (__bf16*)smem;

  int tid = threadIdx.x;
  int lane = tid & 31, wave = tid >> 5;
  int nGroups = N >> 7;                 // (N/16)/8
  int tm = blockIdx.x / nGroups;
  int tnb = blockIdx.x - tm * nGroups;
  int rowBase = tm << 4;
  int KT = K1 + K2;

  // ---- cooperative stage: A stripe (16 x KT) f32 -> bf16 in LDS ----
  for (int r = 0; r < 16; ++r) {
    int gr = rowBase + r;
    const float* src1;
    if (a1_swz) {
      // time-major logical row gr = s*64+b over t(B=64,S=256,D=512)
      int b = gr & 63, s = gr >> 6;
      src1 = A1 + ((size_t)b * 256 + s) * 512;
    } else {
      src1 = A1 + (size_t)gr * lda1;
    }
    for (int col = tid; col < K1; col += 256) aS[r * KT + col] = (__bf16)src1[col];
    if (A2) {
      const float* src2 = A2 + (size_t)gr * lda2;
      for (int col = tid; col < K2; col += 256) aS[r * KT + K1 + col] = (__bf16)src2[col];
    }
  }
  __syncthreads();

  // ---- WMMA chains (one per weight source; no per-iteration selects) ----
  int colBase = ((tnb << 3) + wave) << 4;
  v8f acc = {0.f, 0.f, 0.f, 0.f, 0.f, 0.f, 0.f, 0.f};
  acc = k_chain(acc, aS, KT, 0, K1, W1, ldw1, colBase, lane);
  if (A2) acc = k_chain(acc, aS, KT, K1, K2, W2, ldw2, colBase, lane);

  // ---- epilogue (C/D layout: element r <-> M = r + 8*(lane>=16), N = lane&15) ----
  int n = colBase + (lane & 15);
  float bv = bias ? bias[n] : 0.0f;
  int mOff = (lane >> 4) * 8;
#pragma unroll
  for (int r = 0; r < 8; ++r) {
    int m = rowBase + mOff + r;
    float v = acc[r] + bv;
    if (Yadd) v += Yadd[(size_t)m * ldadd + n];
    if (act == 1) v = tanhf(v);
    Y[(size_t)m * ldy + n] = v;
    if (Y2) Y2[(size_t)m * ldy2 + n] = v;
  }
}

// ---- Persistent bidirectional-LSTM recurrence (encoder, H=64) ---------------
// grid = 2 blocks (dir), 256 threads (8 waves). h,c,gates live in LDS.
// pre* = precomputed x@Wih^T + b, time-major (S*B, 256). Whh bf16 (256,64).
// mode 0: write per-step h into x1(S,B,128) at col dir*64. mode 1: final h -> ht(64,128).
__global__ void k_enc_rnn(const float* __restrict__ preF, const float* __restrict__ preB,
                          const bf16_t* __restrict__ WhhF, const bf16_t* __restrict__ WhhB,
                          float* __restrict__ x1, float* __restrict__ ht, int mode) {
  extern __shared__ float lds[];
  float* hS = lds;            // 64*64
  float* cS = lds + 4096;     // 64*64
  float* gS = lds + 8192;     // 64*256
  int dir = blockIdx.x;
  const float* pre = dir ? preB : preF;
  const bf16_t* Whh = dir ? WhhB : WhhF;
  int tid = threadIdx.x;
  int lane = tid & 31, wave = tid >> 5;

  for (int i = tid; i < 4096; i += 256) { hS[i] = 0.f; cS[i] = 0.f; }
  __syncthreads();

  for (int it = 0; it < 256; ++it) {
    int step = dir ? (255 - it) : it;
    // gates = h @ Whh^T + pre[step]   (64x256, K=64) -> 64 tiles, 8 per wave
#pragma unroll
    for (int t8 = 0; t8 < 8; ++t8) {
      int tile = wave * 8 + t8;
      int tm = tile >> 4, tn = tile & 15;
      v8f acc = {0.f, 0.f, 0.f, 0.f, 0.f, 0.f, 0.f, 0.f};
      int r = (tm << 4) + (lane & 15);
      int koff = (lane >> 4) * 16;
      v16bf b0 = load_b(Whh, 64, tn << 4, 0, lane);
      v16bf b1 = load_b(Whh, 64, tn << 4, 32, lane);
      v16bf a0, a1;
#pragma unroll
      for (int j = 0; j < 8; ++j) a0[j] = (__bf16)hS[r * 64 + koff + j];
#pragma unroll
      for (int j = 0; j < 8; ++j) a0[8 + j] = (__bf16)hS[r * 64 + koff + 8 + j];
#pragma unroll
      for (int j = 0; j < 8; ++j) a1[j] = (__bf16)hS[r * 64 + 32 + koff + j];
#pragma unroll
      for (int j = 0; j < 8; ++j) a1[8 + j] = (__bf16)hS[r * 64 + 32 + koff + 8 + j];
      acc = __builtin_amdgcn_wmma_f32_16x16x32_bf16(false, a0, false, b0, (short)0, acc, false, false);
      acc = __builtin_amdgcn_wmma_f32_16x16x32_bf16(false, a1, false, b1, (short)0, acc, false, false);
      int n = (tn << 4) + (lane & 15);
      int mOff = (lane >> 4) * 8;
#pragma unroll
      for (int r8 = 0; r8 < 8; ++r8) {
        int m = (tm << 4) + mOff + r8;
        gS[m * 256 + n] = acc[r8] + pre[((size_t)step * 64 + m) * 256 + n];
      }
    }
    __syncthreads();
    // elementwise LSTM cell (gate order i|f|g|o)
    for (int i = tid; i < 4096; i += 256) {
      int b = i >> 6, j = i & 63;
      float gi = gS[b * 256 + j];
      float gf = gS[b * 256 + 64 + j];
      float gg = gS[b * 256 + 128 + j];
      float go = gS[b * 256 + 192 + j];
      float cc = sigf(gf) * cS[i] + sigf(gi) * tanhf(gg);
      float hh = sigf(go) * tanhf(cc);
      cS[i] = cc; hS[i] = hh;
      if (mode == 0) x1[((size_t)step * 64 + b) * 128 + dir * 64 + j] = hh;
    }
    __syncthreads();
  }
  if (mode == 1) {
    for (int i = tid; i < 4096; i += 256) {
      int b = i >> 6, j = i & 63;
      ht[b * 128 + dir * 64 + j] = hS[i];
    }
  }
}

// ---- elementwise LSTM cell over (64,H) with strided views -------------------
__global__ void k_lstm_ew(const float* __restrict__ G, int ldg,
                          float* __restrict__ h, int ldh,
                          float* __restrict__ c, int ldc,
                          float* __restrict__ hx, int ldx, int H) {
  int i = blockIdx.x * blockDim.x + threadIdx.x;
  if (i >= 64 * H) return;
  int b = i / H, j = i - b * H;
  const float* g = G + (size_t)b * ldg;
  float gi = g[j], gf = g[H + j], gg = g[2 * H + j], go = g[3 * H + j];
  float cc = sigf(gf) * c[(size_t)b * ldc + j] + sigf(gi) * tanhf(gg);
  float hh = sigf(go) * tanhf(cc);
  c[(size_t)b * ldc + j] = cc;
  h[(size_t)b * ldh + j] = hh;
  if (hx) hx[(size_t)b * ldx + j] = hh;
}

// ---- latent fuse: zstd = softplus(pre); z = mean + eps*zstd -> xcat[:, :512]
__global__ void k_zfuse(const float* __restrict__ zmean, const float* __restrict__ zpre,
                        const float* __restrict__ eps, float* __restrict__ zstd_out,
                        float* __restrict__ xcat) {
  int i = blockIdx.x * blockDim.x + threadIdx.x;
  if (i >= 64 * 512) return;
  int b = i >> 9, j = i & 511;
  float x = zpre[i];
  float sp = (x > 20.f) ? x : log1pf(__expf(x));
  zstd_out[i] = sp;
  xcat[(size_t)b * 1024 + j] = zmean[i] + eps[i] * sp;
}

__global__ void k_zero(float* __restrict__ p, int n) {
  int i = blockIdx.x * blockDim.x + threadIdx.x;
  if (i < n) p[i] = 0.f;
}

__global__ void k_cvt(const float* __restrict__ s, bf16_t* __restrict__ d, int n) {
  int i = blockIdx.x * blockDim.x + threadIdx.x;
  if (i < n) d[i] = (__bf16)s[i];
}

// ---- in-place softmax over rows of 512 (16384 rows) -------------------------
__global__ void k_softmax(float* __restrict__ out) {
  __shared__ float red[256];
  int row = blockIdx.x;
  float* p = out + (size_t)row * 512;
  int tid = threadIdx.x;
  float v0 = p[tid], v1 = p[tid + 256];
  red[tid] = fmaxf(v0, v1);
  __syncthreads();
  for (int s = 128; s > 0; s >>= 1) {
    if (tid < s) red[tid] = fmaxf(red[tid], red[tid + s]);
    __syncthreads();
  }
  float mx = red[0];
  __syncthreads();
  float e0 = __expf(v0 - mx), e1 = __expf(v1 - mx);
  red[tid] = e0 + e1;
  __syncthreads();
  for (int s = 128; s > 0; s >>= 1) {
    if (tid < s) red[tid] += red[tid + s];
    __syncthreads();
  }
  float inv = 1.0f / red[0];
  p[tid] = e0 * inv;
  p[tid + 256] = e1 * inv;
}

// ============================================================================
extern "C" void kernel_launch(void* const* d_in, const int* in_sizes, int n_in,
                              void* d_out, int out_size, void* d_ws, size_t ws_size,
                              hipStream_t stream) {
  const int B = 64, S = 256, D = 512, LAT = 512;
  const float* t       = (const float*)d_in[0];
  const float* seg_emb = (const float*)d_in[1];
  const float* eps     = (const float*)d_in[2];
  const float* enc_b_0f = (const float*)d_in[5];
  const float* enc_b_0b = (const float*)d_in[8];
  const float* enc_b_1f = (const float*)d_in[11];
  const float* enc_b_1b = (const float*)d_in[14];
  const float* fc_mean_b = (const float*)d_in[16];
  const float* fc_std_b  = (const float*)d_in[18];
  const float* fc2_b   = (const float*)d_in[21];
  const float* cond_b_0 = (const float*)d_in[24];
  const float* cond_b_1 = (const float*)d_in[27];
  const float* fc3_b   = (const float*)d_in[29];
  const float* cell1_b = (const float*)d_in[32];
  const float* cell2_b = (const float*)d_in[35];
  const float* fc4_b   = (const float*)d_in[37];

  float* out_main = (float*)d_out;                 // (B,S,D)
  float* zmean_o  = out_main + (size_t)B * S * D;  // (B,512)
  float* zstd_o   = zmean_o + (size_t)B * LAT;     // (B,512)

  // ---- workspace carving ----
  char* base = (char*)d_ws;
  size_t off = 0;
  auto alloc = [&](size_t bytes) -> void* {
    off = (off + 255) & ~(size_t)255;
    void* p = base + off;
    off += bytes;
    return p;
  };
  // bf16 weights (~82 MB total -> resident in the 192 MB L2)
  bf16_t* wWih0f = (bf16_t*)alloc(131072 * 2);
  bf16_t* wWhh0f = (bf16_t*)alloc(16384 * 2);
  bf16_t* wWih0b = (bf16_t*)alloc(131072 * 2);
  bf16_t* wWhh0b = (bf16_t*)alloc(16384 * 2);
  bf16_t* wWih1f = (bf16_t*)alloc(32768 * 2);
  bf16_t* wWhh1f = (bf16_t*)alloc(16384 * 2);
  bf16_t* wWih1b = (bf16_t*)alloc(32768 * 2);
  bf16_t* wWhh1b = (bf16_t*)alloc(16384 * 2);
  bf16_t* wMean  = (bf16_t*)alloc(65536 * 2);
  bf16_t* wStd   = (bf16_t*)alloc(65536 * 2);
  bf16_t* wSeg   = (bf16_t*)alloc(65536 * 2);
  bf16_t* wFc2   = (bf16_t*)alloc((size_t)4194304 * 2);
  bf16_t* wCWhh0 = (bf16_t*)alloc((size_t)4194304 * 2);
  bf16_t* wCWih1 = (bf16_t*)alloc((size_t)4194304 * 2);
  bf16_t* wCWhh1 = (bf16_t*)alloc((size_t)4194304 * 2);
  bf16_t* wFc3   = (bf16_t*)alloc((size_t)4194304 * 2);
  bf16_t* wC1Wih = (bf16_t*)alloc((size_t)6291456 * 2);
  bf16_t* wC1Whh = (bf16_t*)alloc((size_t)4194304 * 2);
  bf16_t* wC2Wih = (bf16_t*)alloc((size_t)4194304 * 2);
  bf16_t* wC2Whh = (bf16_t*)alloc((size_t)4194304 * 2);
  bf16_t* wFc4   = (bf16_t*)alloc(524288 * 2);
  // f32 activations
  float* gates0f = (float*)alloc((size_t)4194304 * 4);  // reused for layer1 fwd pre-gates
  float* gates0b = (float*)alloc((size_t)4194304 * 4);  // reused for layer1 bwd pre-gates
  float* x1      = (float*)alloc((size_t)2097152 * 4);
  float* ht      = (float*)alloc(8192 * 4);
  float* zpre    = (float*)alloc(32768 * 4);
  float* xcat    = (float*)alloc(65536 * 4);
  float* ttbuf   = (float*)alloc(262144 * 4);   // conductor h0|h1|c0|c1 slices
  float* embs    = (float*)alloc((size_t)1048576 * 4);
  float* cgates  = (float*)alloc(262144 * 4);
  float* gbuf    = (float*)alloc(262144 * 4);   // decoder dh1|dh2|dc1|dc2 slices
  float* embWih  = (float*)alloc(262144 * 4);
  float* dgates  = (float*)alloc(262144 * 4);
  float* prevbuf = (float*)alloc(32768 * 4);

  // ---- weight f32 -> bf16 conversion ----
  struct Cvt { int idx; int n; bf16_t* dst; };
  const Cvt cv[] = {
    {3, 131072, wWih0f}, {4, 16384, wWhh0f}, {6, 131072, wWih0b}, {7, 16384, wWhh0b},
    {9, 32768, wWih1f}, {10, 16384, wWhh1f}, {12, 32768, wWih1b}, {13, 16384, wWhh1b},
    {15, 65536, wMean}, {17, 65536, wStd}, {19, 65536, wSeg},
    {20, 4194304, wFc2}, {23, 4194304, wCWhh0}, {25, 4194304, wCWih1}, {26, 4194304, wCWhh1},
    {28, 4194304, wFc3}, {30, 6291456, wC1Wih}, {31, 4194304, wC1Whh},
    {33, 4194304, wC2Wih}, {34, 4194304, wC2Whh}, {36, 524288, wFc4},
  };
  for (const Cvt& c : cv)
    k_cvt<<<(c.n + 255) / 256, 256, 0, stream>>>((const float*)d_in[c.idx], c.dst, c.n);

  auto gemm = [&](float* Y, int ldy, float* Y2, long long ldy2,
                  const float* Yadd, int ldadd,
                  const float* A1, int lda1, int K1, const bf16_t* W1, int ldw1,
                  const float* A2, int lda2, int K2, const bf16_t* W2, int ldw2,
                  const float* bias, int M, int N, int act, int swz) {
    int blocks = (M / 16) * (N / 128);
    size_t shmem = (size_t)16 * (K1 + K2) * sizeof(bf16_t);
    k_gemm<<<blocks, 256, shmem, stream>>>(Y, ldy, Y2, ldy2, Yadd, ldadd,
                                           A1, lda1, K1, W1, ldw1, A2, lda2, K2, W2, ldw2,
                                           bias, N, act, swz);
  };
  float* NF = nullptr;
  const bf16_t* NW = nullptr;

  // ---- encoder: hoisted input projections (big WMMA GEMMs), then persistent RNN
  gemm(gates0f, 256, NF, 0, NF, 0, t, 512, 512, wWih0f, 512, NF, 0, 0, NW, 0,
       enc_b_0f, S * B, 256, 0, 1);
  gemm(gates0b, 256, NF, 0, NF, 0, t, 512, 512, wWih0b, 512, NF, 0, 0, NW, 0,
       enc_b_0b, S * B, 256, 0, 1);
  k_enc_rnn<<<2, 256, 98304, stream>>>(gates0f, gates0b, wWhh0f, wWhh0b, x1, NF, 0);
  gemm(gates0f, 256, NF, 0, NF, 0, x1, 128, 128, wWih1f, 128, NF, 0, 0, NW, 0,
       enc_b_1f, S * B, 256, 0, 0);
  gemm(gates0b, 256, NF, 0, NF, 0, x1, 128, 128, wWih1b, 128, NF, 0, 0, NW, 0,
       enc_b_1b, S * B, 256, 0, 0);
  k_enc_rnn<<<2, 256, 98304, stream>>>(gates0f, gates0b, wWhh1f, wWhh1b, NF, ht, 1);

  // ---- latent ----
  gemm(zmean_o, 512, NF, 0, NF, 0, ht, 128, 128, wMean, 128, NF, 0, 0, NW, 0,
       fc_mean_b, 64, 512, 0, 0);
  gemm(zpre, 512, NF, 0, NF, 0, ht, 128, 128, wStd, 128, NF, 0, 0, NW, 0,
       fc_std_b, 64, 512, 0, 0);
  gemm(xcat + 512, 1024, NF, 0, NF, 0, seg_emb, 128, 128, wSeg, 128, NF, 0, 0, NW, 0,
       nullptr, 64, 512, 0, 0);
  k_zfuse<<<(64 * 512 + 255) / 256, 256, 0, stream>>>(zmean_o, zpre, eps, zstd_o, xcat);

  // ---- conductor init: tt = tanh(xcat @ fc2^T + b), slices h0|h1|c0|c1 ----
  gemm(ttbuf, 4096, NF, 0, NF, 0, xcat, 1024, 1024, wFc2, 1024, NF, 0, 0, NW, 0,
       fc2_b, 64, 4096, 1, 0);
  float* ch0 = ttbuf;          // h layer0
  float* ch1 = ttbuf + 1024;   // h layer1
  float* cc0 = ttbuf + 2048;   // c layer0
  float* cc1 = ttbuf + 3072;   // c layer1

  // ---- conductor: 16 steps, layer0 input is zero (Wih term drops) ----
  for (int step = 0; step < 16; ++step) {
    gemm(cgates, 4096, NF, 0, NF, 0, ch0, 4096, 1024, wCWhh0, 1024, NF, 0, 0, NW, 0,
         cond_b_0, 64, 4096, 0, 0);
    k_lstm_ew<<<256, 256, 0, stream>>>(cgates, 4096, ch0, 4096, cc0, 4096, NF, 0, 1024);
    gemm(cgates, 4096, NF, 0, NF, 0, ch0, 4096, 1024, wCWih1, 1024,
         ch1, 4096, 1024, wCWhh1, 1024, cond_b_1, 64, 4096, 0, 0);
    k_lstm_ew<<<256, 256, 0, stream>>>(cgates, 4096, ch1, 4096, cc1, 4096,
                                       embs + (size_t)step * 65536, 1024, 1024);
  }

  // ---- decoder: 16 bars x 16 steps, fully sequential ----
  k_zero<<<(32768 + 255) / 256, 256, 0, stream>>>(prevbuf, 32768);
  float* dh1 = gbuf;
  float* dh2 = gbuf + 1024;
  float* dc1 = gbuf + 2048;
  float* dc2 = gbuf + 3072;
  for (int bar = 0; bar < 16; ++bar) {
    const float* emb = embs + (size_t)bar * 65536;
    // g = tanh(emb @ fc3^T + b) -> dh1|dh2|dc1|dc2 slices of gbuf
    gemm(gbuf, 4096, NF, 0, NF, 0, emb, 1024, 1024, wFc3, 1024, NF, 0, 0, NW, 0,
         fc3_b, 64, 4096, 1, 0);
    // per-bar constant part of cell1 gates: emb @ Wih[:, :1024]^T + cell1_b
    gemm(embWih, 4096, NF, 0, NF, 0, emb, 1024, 1024, wC1Wih, 1536, NF, 0, 0, NW, 0,
         cell1_b, 64, 4096, 0, 0);
    for (int s = 0; s < 16; ++s) {
      // cell1 gates = prev @ Wih[:,1024:]^T + dh1 @ Whh^T + embWih
      gemm(dgates, 4096, NF, 0, embWih, 4096,
           prevbuf, 512, 512, wC1Wih + 1024, 1536,
           dh1, 4096, 1024, wC1Whh, 1024, nullptr, 64, 4096, 0, 0);
      k_lstm_ew<<<256, 256, 0, stream>>>(dgates, 4096, dh1, 4096, dc1, 4096, NF, 0, 1024);
      // cell2 gates = dh1 @ Wih^T + dh2 @ Whh^T + b
      gemm(dgates, 4096, NF, 0, NF, 0,
           dh1, 4096, 1024, wC2Wih, 1024,
           dh2, 4096, 1024, wC2Whh, 1024, cell2_b, 64, 4096, 0, 0);
      k_lstm_ew<<<256, 256, 0, stream>>>(dgates, 4096, dh2, 4096, dc2, 4096, NF, 0, 1024);
      // prev = dh2 @ fc4^T + b ; also scatter into out[:, bar*16+s, :]
      int sg = bar * 16 + s;
      gemm(prevbuf, 512, out_main + (size_t)sg * 512, (long long)S * D, NF, 0,
           dh2, 4096, 1024, wFc4, 1024, NF, 0, 0, NW, 0,
           fc4_b, 64, 512, 0, 0);
    }
  }

  // ---- softmax over D in-place ----
  k_softmax<<<B * S, 256, 0, stream>>>(out_main);
}
